// Attention_28046136443022
// MI455X (gfx1250) — matrix operand order repaired and checked by reading, not compiled, not run on hardware
//
#include <hip/hip_runtime.h>
#include <hip/hip_bf16.h>
#include <math.h>

// ---------------------------------------------------------------------------
// MHA forward on gfx1250: bf16 WMMA (v_wmma_f32_16x16x32_bf16), f32 accumulate.
// [qkv gemm: 128x128 tiles, double-buffered LDS]
// [flash attention: block-shared K/V via global_load_async_to_lds_b128]
// [out-proj gemm: 128x128 tiles, double-buffered LDS]
// ---------------------------------------------------------------------------

typedef __attribute__((ext_vector_type(16))) __bf16 v16bf;
typedef __attribute__((ext_vector_type(8)))  __bf16 v8bf;
typedef __attribute__((ext_vector_type(8)))  float  v8f;

#define NUM_HEADS 12
#define HD   64
#define SEQ  1024
#define DM   768
#define BATCH 8
#define ROWS (BATCH*SEQ)   // 8192
#define QKVN (3*DM)        // 2304
#define LSTRIDE 40         // padded LDS stride (16B-aligned frags, bank spread)
#define KSTRIDE 72         // padded K-tile row stride

static __device__ __forceinline__ __bf16 to_bf16(float f) { return (__bf16)f; }

// A operand, 16x32 bf16 (ISA 7.12.2): lane l -> row l&15;
// elements [0..7] = K(8*ksel+0..7), [8..15] = K(16+8*ksel+0..7), ksel = l>>4.
static __device__ __forceinline__ v16bf ld_frag_a(const __bf16* base, int lane, int stride) {
    const int r = lane & 15, ksel = (lane >> 4) & 1;
    const __bf16* p = base + r * stride + 8 * ksel;
    v8bf lo = *(const v8bf*)(p);
    v8bf hi = *(const v8bf*)(p + 16);
    v16bf a;
#pragma unroll
    for (int i = 0; i < 8; ++i) { a[i] = lo[i]; a[i + 8] = hi[i]; }
    return a;
}

// B operand, 32x16 bf16: lane l -> column l&15; elements [0..7] = K(8*kh+0..7), [8..15] = K(16+8*kh+0..7), kh = l>>4.
static __device__ __forceinline__ v16bf ld_frag_b(const __bf16* base, int lane, int stride) {
    const int c = lane & 15, kh = (lane >> 4) & 1;
    const __bf16* p = base + (size_t)c * stride + 8 * kh;
    v8bf lo = *(const v8bf*)(p);
    v8bf hi = *(const v8bf*)(p + 16);
    v16bf b;
#pragma unroll
    for (int i = 0; i < 8; ++i) { b[i] = lo[i]; b[i + 8] = hi[i]; }
    return b;
}

static __device__ __forceinline__ v8f wmma_bf16(v16bf a, v16bf b, v8f c) {
    return __builtin_amdgcn_wmma_f32_16x16x32_bf16(false, a, false, b, (short)0, c, false, false);
}

// CDNA5 async copy: 16B global -> LDS, tracked by ASYNCcnt (in-order).
static __device__ __forceinline__ void async_copy_b128(const void* gptr, void* lptr) {
    unsigned lds_off = (unsigned)(uintptr_t)lptr;           // low 32 bits = LDS offset
    unsigned long long gaddr = (unsigned long long)(uintptr_t)gptr;
    asm volatile("global_load_async_to_lds_b128 %0, %1, off"
                 :: "v"(lds_off), "v"(gaddr) : "memory");
}
static __device__ __forceinline__ void wait_async_le2() {
    asm volatile("s_wait_asynccnt 0x2" ::: "memory");
}
static __device__ __forceinline__ void wait_async_0() {
    asm volatile("s_wait_asynccnt 0x0" ::: "memory");
}

// ---------------------------------------------------------------------------
// Kernel 1: QKV = x @ w_qkv + b_qkv ; scatter to Q[B,H,N,hd], K[B,H,N,hd],
// Vt[B,H,hd,N] bf16. Tile 128(M) x 128(N), 8 waves of 32x64, K-step 32,
// double-buffered LDS, deserialized staging.
// ---------------------------------------------------------------------------
__global__ __launch_bounds__(256)
void qkv_gemm_kernel(const float* __restrict__ x, const float* __restrict__ w,
                     const float* __restrict__ bias,
                     __bf16* __restrict__ Q, __bf16* __restrict__ K,
                     __bf16* __restrict__ Vt)
{
    __shared__ __bf16 ldsA[2][128 * LSTRIDE];   // row-major rows of x tile
    __shared__ __bf16 ldsB[2][128 * LSTRIDE];   // column-major cols of w tile
    const int tid = threadIdx.x;
    const int lane = tid & 31, wave = tid >> 5;
    const int wm = wave & 3, wn = wave >> 2;    // 4x2 wave grid -> 32x64 per wave
    const int row0 = blockIdx.y * 128;
    const int col0 = blockIdx.x * 128;

    float4 ra[4], rb[4];
    // phase 1: issue all 8 global loads into distinct registers (MLP!)
    auto loadR = [&](int kb) {
#pragma unroll
        for (int it = 0; it < 4; ++it) {
            int lin = tid + it * 256;
            ra[it] = *(const float4*)(x + (size_t)(row0 + (lin >> 3)) * DM + kb + (lin & 7) * 4);
        }
#pragma unroll
        for (int it = 0; it < 4; ++it) {
            int lin = tid + it * 256;
            rb[it] = *(const float4*)(w + (size_t)(kb + (lin >> 5)) * QKVN + col0 + (lin & 31) * 4);
        }
    };
    // phase 2: convert fp32 -> bf16 and store to LDS
    auto storeL = [&](int buf) {
#pragma unroll
        for (int it = 0; it < 4; ++it) {
            int lin = tid + it * 256;
            __bf16* d = &ldsA[buf][(lin >> 3) * LSTRIDE + (lin & 7) * 4];
            d[0] = to_bf16(ra[it].x); d[1] = to_bf16(ra[it].y);
            d[2] = to_bf16(ra[it].z); d[3] = to_bf16(ra[it].w);
        }
#pragma unroll
        for (int it = 0; it < 4; ++it) {
            int lin = tid + it * 256;
            int kr = lin >> 5, c4 = (lin & 31) * 4;
            ldsB[buf][(c4 + 0) * LSTRIDE + kr] = to_bf16(rb[it].x);
            ldsB[buf][(c4 + 1) * LSTRIDE + kr] = to_bf16(rb[it].y);
            ldsB[buf][(c4 + 2) * LSTRIDE + kr] = to_bf16(rb[it].z);
            ldsB[buf][(c4 + 3) * LSTRIDE + kr] = to_bf16(rb[it].w);
        }
    };

    v8f acc[2][4] = {};
    loadR(0);
    storeL(0);
    const int NT = DM / 32;
    for (int t = 0; t < NT; ++t) {
        __syncthreads();
        if (t + 1 < NT) loadR((t + 1) * 32);   // next tile fetch overlaps WMMAs
        const int cur = t & 1;
        v16bf af[2], bfv[4];
#pragma unroll
        for (int ms = 0; ms < 2; ++ms)
            af[ms] = ld_frag_a(&ldsA[cur][(wm * 32 + ms * 16) * LSTRIDE], lane, LSTRIDE);
#pragma unroll
        for (int ns = 0; ns < 4; ++ns)
            bfv[ns] = ld_frag_b(&ldsB[cur][(wn * 64 + ns * 16) * LSTRIDE], lane, LSTRIDE);
#pragma unroll
        for (int ms = 0; ms < 2; ++ms)
#pragma unroll
            for (int ns = 0; ns < 4; ++ns)
                acc[ms][ns] = wmma_bf16(af[ms], bfv[ns], acc[ms][ns]);
        if (t + 1 < NT) storeL(cur ^ 1);
    }

    // epilogue: bias + scatter (C layout: lane -> col lane&15, rows i+8*(lane>>4))
    const int hrow = lane >> 4;
#pragma unroll
    for (int ms = 0; ms < 2; ++ms) {
#pragma unroll
        for (int ns = 0; ns < 4; ++ns) {
            int j = col0 + wn * 64 + ns * 16 + (lane & 15);
            float bv = bias[j];
            int s = j / DM, rr = j % DM;
            int h = rr / HD, d = rr % HD;
#pragma unroll
            for (int i = 0; i < 8; ++i) {
                int gm = row0 + wm * 32 + ms * 16 + i + 8 * hrow;
                int bidx = gm >> 10, n = gm & (SEQ - 1);
                float val = acc[ms][ns][i] + bv;
                size_t bh = (size_t)(bidx * NUM_HEADS + h);
                if (s == 0)      Q [(bh * SEQ + n) * HD + d] = to_bf16(val);
                else if (s == 1) K [(bh * SEQ + n) * HD + d] = to_bf16(val);
                else             Vt[(bh * HD + d) * SEQ + n] = to_bf16(val);
            }
        }
    }
}

// ---------------------------------------------------------------------------
// Kernel 2: flash attention. Block = 8 waves = 128 queries of one (b,h).
// K/V key-blocks staged block-wide in LDS via async DMA, double-buffered.
// ---------------------------------------------------------------------------
__global__ __launch_bounds__(256)
void attention_kernel(const __bf16* __restrict__ Q, const __bf16* __restrict__ K,
                      const __bf16* __restrict__ Vt, __bf16* __restrict__ A)
{
    __shared__ __bf16 ldsK[2][32 * KSTRIDE];     // [key][hd], padded rows
    __shared__ __bf16 ldsV[2][HD * LSTRIDE];     // [d][key], padded rows
    __shared__ __bf16 ldsP[8][16 * LSTRIDE];     // per-wave P tile (C->A relayout)

    const int tid = threadIdx.x, lane = tid & 31, wave = tid >> 5;
    const int blk = blockIdx.x;
    const int bh = blk >> 3;                     // b*H + h
    const int qt = blk & 7;
    const int b = bh / NUM_HEADS, h = bh % NUM_HEADS;
    const int q0 = qt * 128 + wave * 16;
    const float scale = 0.125f;                  // 1/sqrt(64)

    const __bf16* Qp = Q + ((size_t)bh * SEQ + q0) * HD;
    const __bf16* Kp = K + (size_t)bh * SEQ * HD;
    const __bf16* Vp = Vt + (size_t)bh * HD * SEQ;

    // per-thread 16B chunks: K tile 32x64 (256 chunks), V tile 64x32 (256 chunks)
    const int k_key = tid >> 3, k_ch = tid & 7;  // key row, 8B-elem chunk
    const int v_d = tid >> 2, v_ch = tid & 3;
    auto issue_async = [&](int buf, int kb) {
        async_copy_b128(Kp + (size_t)(kb + k_key) * HD + k_ch * 8,
                        &ldsK[buf][k_key * KSTRIDE + k_ch * 8]);
        async_copy_b128(Vp + (size_t)v_d * SEQ + kb + v_ch * 8,
                        &ldsV[buf][v_d * LSTRIDE + v_ch * 8]);
    };

    v16bf qf[2];
    qf[0] = ld_frag_a(Qp, lane, HD);             // hd 0..31
    qf[1] = ld_frag_a(Qp + 32, lane, HD);        // hd 32..63

    v8f o[4] = {};
    float mrun[8], lrun[8];
#pragma unroll
    for (int i = 0; i < 8; ++i) { mrun[i] = -1e30f; lrun[i] = 0.f; }

    issue_async(0, 0);
    const int NT = SEQ / 32;
    for (int t = 0; t < NT; ++t) {
        const int cur = t & 1;
        if (t + 1 < NT) { issue_async(cur ^ 1, (t + 1) * 32); wait_async_le2(); }
        else            { wait_async_0(); }
        __syncthreads();                         // all waves' copies of cur visible

        // S tile: 16 queries x 32 keys, contraction over hd in two WMMA steps
        v8f s[2] = {};
#pragma unroll
        for (int kg = 0; kg < 2; ++kg) {
            const __bf16* kbase = &ldsK[cur][kg * 16 * KSTRIDE];
            v16bf b0 = ld_frag_b(kbase,      lane, KSTRIDE);
            v16bf b1 = ld_frag_b(kbase + 32, lane, KSTRIDE);
            s[kg] = wmma_bf16(qf[0], b0, s[kg]);
            s[kg] = wmma_bf16(qf[1], b1, s[kg]);
        }

        // online softmax: rows i+8*(lane>>4); 16 key-cols live in 16 lanes
        float p0[8], p1[8], cor[8];
#pragma unroll
        for (int i = 0; i < 8; ++i) {
            float s0 = s[0][i] * scale, s1 = s[1][i] * scale;
            float rm = fmaxf(s0, s1);
#pragma unroll
            for (int off = 1; off < 16; off <<= 1)
                rm = fmaxf(rm, __shfl_xor(rm, off, 32));
            float mnew = fmaxf(mrun[i], rm);
            cor[i] = __expf(mrun[i] - mnew);
            p0[i] = __expf(s0 - mnew);
            p1[i] = __expf(s1 - mnew);
            float r = p0[i] + p1[i];
#pragma unroll
            for (int off = 1; off < 16; off <<= 1)
                r += __shfl_xor(r, off, 32);
            lrun[i] = lrun[i] * cor[i] + r;
            mrun[i] = mnew;
        }
#pragma unroll
        for (int tt = 0; tt < 4; ++tt)
#pragma unroll
            for (int i = 0; i < 8; ++i) o[tt][i] *= cor[i];

        // C-layout -> A-layout via per-wave LDS tile (DS in-order per wave)
        __bf16* Pl = &ldsP[wave][0];
        {
            int n = lane & 15, hh = lane >> 4;
#pragma unroll
            for (int i = 0; i < 8; ++i) {
                int m = i + 8 * hh;
                Pl[m * LSTRIDE + n]      = to_bf16(p0[i]);
                Pl[m * LSTRIDE + 16 + n] = to_bf16(p1[i]);
            }
        }
        __builtin_amdgcn_wave_barrier();
        asm volatile("" ::: "memory");

        v16bf pf = ld_frag_a(Pl, lane, LSTRIDE);
#pragma unroll
        for (int tt = 0; tt < 4; ++tt) {
            v16bf vf = ld_frag_b(&ldsV[cur][tt * 16 * LSTRIDE], lane, LSTRIDE);
            o[tt] = wmma_bf16(pf, vf, o[tt]);
        }
        __syncthreads();   // everyone done reading cur before it is overwritten
    }

    // normalize and store attention output as bf16 rows of [B*N, 768]
    const int hh = lane >> 4, n = lane & 15;
#pragma unroll
    for (int tt = 0; tt < 4; ++tt) {
#pragma unroll
        for (int i = 0; i < 8; ++i) {
            int m = i + 8 * hh;
            float val = o[tt][i] / lrun[i];
            size_t row = (size_t)b * SEQ + q0 + m;
            A[row * DM + h * HD + tt * 16 + n] = to_bf16(val);
        }
    }
}

// ---------------------------------------------------------------------------
// Kernel 3: out = A @ w_proj + b_proj (A already bf16), fp32 output.
// Same 128x128 double-buffered scheme.
// ---------------------------------------------------------------------------
__global__ __launch_bounds__(256)
void proj_gemm_kernel(const __bf16* __restrict__ A, const float* __restrict__ w,
                      const float* __restrict__ bias, float* __restrict__ out)
{
    __shared__ __bf16 ldsA[2][128 * LSTRIDE];
    __shared__ __bf16 ldsB[2][128 * LSTRIDE];
    const int tid = threadIdx.x, lane = tid & 31, wave = tid >> 5;
    const int wm = wave & 3, wn = wave >> 2;
    const int row0 = blockIdx.y * 128;
    const int col0 = blockIdx.x * 128;

    v8bf ra[2];
    float4 rb[4];
    auto loadR = [&](int kb) {
#pragma unroll
        for (int it = 0; it < 2; ++it) {
            int lin = tid + it * 256;
            ra[it] = *(const v8bf*)(A + (size_t)(row0 + (lin >> 2)) * DM + kb + (lin & 3) * 8);
        }
#pragma unroll
        for (int it = 0; it < 4; ++it) {
            int lin = tid + it * 256;
            rb[it] = *(const float4*)(w + (size_t)(kb + (lin >> 5)) * DM + col0 + (lin & 31) * 4);
        }
    };
    auto storeL = [&](int buf) {
#pragma unroll
        for (int it = 0; it < 2; ++it) {
            int lin = tid + it * 256;
            __bf16* d = &ldsA[buf][(lin >> 2) * LSTRIDE + (lin & 3) * 8];
#pragma unroll
            for (int i = 0; i < 8; ++i) d[i] = ra[it][i];
        }
#pragma unroll
        for (int it = 0; it < 4; ++it) {
            int lin = tid + it * 256;
            int kr = lin >> 5, c4 = (lin & 31) * 4;
            ldsB[buf][(c4 + 0) * LSTRIDE + kr] = to_bf16(rb[it].x);
            ldsB[buf][(c4 + 1) * LSTRIDE + kr] = to_bf16(rb[it].y);
            ldsB[buf][(c4 + 2) * LSTRIDE + kr] = to_bf16(rb[it].z);
            ldsB[buf][(c4 + 3) * LSTRIDE + kr] = to_bf16(rb[it].w);
        }
    };

    v8f acc[2][4] = {};
    loadR(0);
    storeL(0);
    const int NT = DM / 32;
    for (int t = 0; t < NT; ++t) {
        __syncthreads();
        if (t + 1 < NT) loadR((t + 1) * 32);
        const int cur = t & 1;
        v16bf af[2], bfv[4];
#pragma unroll
        for (int ms = 0; ms < 2; ++ms)
            af[ms] = ld_frag_a(&ldsA[cur][(wm * 32 + ms * 16) * LSTRIDE], lane, LSTRIDE);
#pragma unroll
        for (int ns = 0; ns < 4; ++ns)
            bfv[ns] = ld_frag_b(&ldsB[cur][(wn * 64 + ns * 16) * LSTRIDE], lane, LSTRIDE);
#pragma unroll
        for (int ms = 0; ms < 2; ++ms)
#pragma unroll
            for (int ns = 0; ns < 4; ++ns)
                acc[ms][ns] = wmma_bf16(af[ms], bfv[ns], acc[ms][ns]);
        if (t + 1 < NT) storeL(cur ^ 1);
    }

    const int hrow = lane >> 4;
#pragma unroll
    for (int ms = 0; ms < 2; ++ms)
#pragma unroll
        for (int ns = 0; ns < 4; ++ns) {
            int j = col0 + wn * 64 + ns * 16 + (lane & 15);
            float bv = bias[j];
#pragma unroll
            for (int i = 0; i < 8; ++i) {
                int gm = row0 + wm * 32 + ms * 16 + i + 8 * hrow;
                out[(size_t)gm * DM + j] = acc[ms][ns][i] + bv;
            }
        }
}

// ---------------------------------------------------------------------------
extern "C" void kernel_launch(void* const* d_in, const int* in_sizes, int n_in,
                              void* d_out, int out_size, void* d_ws, size_t ws_size,
                              hipStream_t stream)
{
    const float* x      = (const float*)d_in[0];
    const float* w_qkv  = (const float*)d_in[1];
    const float* b_qkv  = (const float*)d_in[2];
    const float* w_proj = (const float*)d_in[3];
    const float* b_proj = (const float*)d_in[4];
    float* out = (float*)d_out;

    char* ws = (char*)d_ws;
    const size_t perT = (size_t)BATCH * NUM_HEADS * SEQ * HD * sizeof(__bf16); // 12.58 MB
    __bf16* Q  = (__bf16*)(ws);
    __bf16* K  = (__bf16*)(ws + perT);
    __bf16* Vt = (__bf16*)(ws + 2 * perT);
    __bf16* Aa = (__bf16*)(ws + 3 * perT);   // [B*N, 768] bf16

    qkv_gemm_kernel<<<dim3(QKVN / 128, ROWS / 128), 256, 0, stream>>>(
        x, w_qkv, b_qkv, Q, K, Vt);
    attention_kernel<<<dim3(BATCH * NUM_HEADS * (SEQ / 128)), 256, 0, stream>>>(
        Q, K, Vt, Aa);
    proj_gemm_kernel<<<dim3(DM / 128, ROWS / 128), 256, 0, stream>>>(
        Aa, w_proj, b_proj, out);
}
